// DCNV2_73452530696362
// MI455X (gfx1250) — compile-verified, matrix operand order checked by reading, not run on hardware
//
#include <hip/hip_runtime.h>
#include <math.h>

// ---------------------------------------------------------------------------
// DCNv2 fused pipeline for gfx1250 (MI455X), wave32 + WMMA bf16.
//   K0: zero BN accumulators
//   K1: swizzle main conv weight   f32[CO][C][3][3] -> bf16 B-fragments
//   K2: swizzle offset/mod weights -> bf16 B-fragments (27 cols padded to 32)
//   K3: offset+mask conv via WMMA (A fragment built in registers, branch-free)
//   K4: fused bilinear-sample + WMMA GEMM + bias, coalesced pre-BN store,
//       per-channel sum/sumsq via f32 atomics
//   K5: BN finalize + ReLU in place on d_out
// ---------------------------------------------------------------------------

typedef __attribute__((ext_vector_type(16))) __bf16 v16bf;
typedef __attribute__((ext_vector_type(8)))  float  v8f;

#define Nb    4
#define Cc    128
#define CO    128
#define Hh    96
#define Ww    96
#define HW    (Hh * Ww)          // 9216
#define KK    9
#define KDIM  (KK * Cc)          // 1152
#define NKT   (KDIM / 32)        // 36 k-tiles of 32
#define Mrows (Nb * HW)          // 36864 output positions
#define BN_EPS 1e-5f

// fragment element -> kl mapping (16-bit A/B layout, ISA 7.12.2):
//   lane l, element e:  kl = ((e<8)? e : e+8) + 8*(l>>4)
__device__ __forceinline__ int frag_kl_of(int g, int e) {
    return ((e < 8) ? e : e + 8) + (g << 3);
}

// ---------------------------------------------------------------- K0: init BN
__global__ void k0_init_bn(float* bnsum, float* bnsumsq) {
    int t = threadIdx.x;
    if (t < CO) { bnsum[t] = 0.f; bnsumsq[t] = 0.f; }
}

// ------------------------------------------------- K1: main weight -> B frags
// wB layout: [ktile(36)][ctile(8)][lane(32)][e(16)] bf16, 32B per lane row.
__global__ void k1_wcvt_main(const float* __restrict__ weight, __bf16* __restrict__ wB) {
    int idx = blockIdx.x * blockDim.x + threadIdx.x;      // 36*8*32 = 9216 lane rows
    if (idx >= NKT * 8 * 32) return;
    int lane  = idx & 31;
    int ctile = (idx >> 5) & 7;
    int ktile = idx >> 8;
    int g = lane >> 4;
    int o = ctile * 16 + (lane & 15);
    __bf16* dst = wB + (size_t)idx * 16;
    #pragma unroll
    for (int e = 0; e < 16; ++e) {
        int k  = ktile * 32 + frag_kl_of(g, e);
        int kk = k >> 7;           // k / 128
        int c  = k & 127;
        dst[e] = (__bf16)weight[((size_t)o * Cc + c) * KK + kk];
    }
}

// -------------------------------- K2: offset(18)+mod(9) weights -> B frags
// wOM layout: [ktile(36)][ctile(2)][lane(32)][e(16)] bf16 (cols 27..31 zero).
__global__ void k2_wcvt_offmask(const float* __restrict__ offw,
                                const float* __restrict__ modw,
                                __bf16* __restrict__ wOM) {
    int idx = blockIdx.x * blockDim.x + threadIdx.x;      // 36*2*32 = 2304 lane rows
    if (idx >= NKT * 2 * 32) return;
    int lane  = idx & 31;
    int ctile = (idx >> 5) & 1;
    int ktile = idx >> 6;
    int g   = lane >> 4;
    int col = ctile * 16 + (lane & 15);
    __bf16* dst = wOM + (size_t)idx * 16;
    #pragma unroll
    for (int e = 0; e < 16; ++e) {
        int k  = ktile * 32 + frag_kl_of(g, e);
        int kk = k >> 7;
        int c  = k & 127;
        float v = 0.f;
        if (col < 18)      v = offw[((size_t)col * Cc + c) * KK + kk];
        else if (col < 27) v = modw[((size_t)(col - 18) * Cc + c) * KK + kk];
        dst[e] = (__bf16)v;
    }
}

// ------------------------- K3: offset/mask conv as WMMA GEMM (1 wave / block)
// block b -> 16 positions (one n,h row segment); cols 0..31 = 18 off + 9 mask.
// A fragment built directly in registers: lane l owns row m = l&15; for a
// fixed element e, lanes 0..15 read 16 consecutive floats (coalesced).
__global__ void k3_offmask_gemm(const float* __restrict__ x,
                                const __bf16* __restrict__ wOM,
                                const float* __restrict__ offb,
                                const float* __restrict__ modb,
                                float* __restrict__ offsBuf,   // [N][18][HW]
                                float* __restrict__ maskBuf)   // [N][9][HW]
{
    const int lane = threadIdx.x;            // 0..31
    const int g    = lane >> 4;
    const int p    = lane & 15;              // position within tile (row m)
    const int m0   = blockIdx.x * 16;
    const int n    = m0 / HW;
    const int rem  = m0 % HW;
    const int h    = rem / Ww;
    const int w0   = rem % Ww;               // W % 16 == 0 -> row-contiguous tile

    v8f acc0 = {}; v8f acc1 = {};
    const float* xn = x + (size_t)n * Cc * HW;

    for (int kk = 0; kk < KK; ++kk) {
        const int dy = kk / 3 - 1, dx = kk % 3 - 1;
        const int yy  = h + dy;
        const int xx  = w0 + p + dx;
        const bool ok = (yy >= 0) && (yy < Hh) && (xx >= 0) && (xx < Ww);
        const float vf = ok ? 1.f : 0.f;
        const int yyc = min(max(yy, 0), Hh - 1);
        const int xxc = min(max(xx, 0), Ww - 1);
        const size_t rowoff = (size_t)yyc * Ww + xxc;
        for (int chunk = 0; chunk < 4; ++chunk) {
            v16bf a;
            #pragma unroll
            for (int e = 0; e < 16; ++e) {
                const int c = chunk * 32 + frag_kl_of(g, e);
                a[e] = (__bf16)(xn[(size_t)c * HW + rowoff] * vf);
            }
            const int ktile = kk * 4 + chunk;
            const v16bf b0 = *reinterpret_cast<const v16bf*>(wOM + ((size_t)(ktile * 2 + 0) * 32 + lane) * 16);
            const v16bf b1 = *reinterpret_cast<const v16bf*>(wOM + ((size_t)(ktile * 2 + 1) * 32 + lane) * 16);
            acc0 = __builtin_amdgcn_wmma_f32_16x16x32_bf16(false, a, false, b0, (short)0, acc0, false, false);
            acc1 = __builtin_amdgcn_wmma_f32_16x16x32_bf16(false, a, false, b1, (short)0, acc1, false, false);
        }
    }

    // epilogue: C/D layout -> lane l, vgpr r: row = r + 8*(l>>4), col = l&15
    const int rowHi = g * 8;
    const int colLo = lane & 15;
    #pragma unroll
    for (int r = 0; r < 8; ++r) {
        const int w = w0 + r + rowHi;
        {   // tile 0: channels 0..15 (all offsets)
            const int ch = colLo;
            offsBuf[((size_t)(n * 18 + ch)) * HW + (size_t)h * Ww + w] = acc0[r] + offb[ch];
        }
        {   // tile 1: channels 16..31
            const int ch = 16 + colLo;
            float v = acc1[r];
            if (ch < 18) {
                offsBuf[((size_t)(n * 18 + ch)) * HW + (size_t)h * Ww + w] = v + offb[ch];
            } else if (ch < 27) {
                const int mc = ch - 18;
                v += modb[mc];
                maskBuf[((size_t)(n * 9 + mc)) * HW + (size_t)h * Ww + w] = 2.f / (1.f + __expf(-v));
            }
        }
    }
}

// ------------------- K4: fused bilinear sample + WMMA main GEMM (4 waves/blk)
// block -> 16 positions x 128 output channels; wave w -> cols 32w..32w+31.
// A tile built cooperatively: thread t owns row l=t>>2, elements 4q..4q+3
// (q=t&3) -> one packed 8-byte LDS store. Bilinear params computed per thread
// in registers (branch-free; validity folded into weights).
__global__ void k4_dcn_gemm(const float* __restrict__ x,
                            const float* __restrict__ offsBuf,
                            const float* __restrict__ maskBuf,
                            const __bf16* __restrict__ wB,
                            const float* __restrict__ bias,
                            float* __restrict__ out,        // pre-BN, NCHW
                            float* __restrict__ bnsum,
                            float* __restrict__ bnsumsq)
{
    __attribute__((aligned(32))) __shared__ __bf16 As[32][16];
    __shared__ float sOut[16][132];           // padded: 16 rows x 128 chans

    const int t    = threadIdx.x;             // 0..127
    const int lane = t & 31;
    const int wav  = t >> 5;                  // 0..3
    const int l    = t >> 2;                  // A row owned by this thread
    const int q    = t & 3;                   // element quarter
    const int g    = l >> 4;
    const int p    = l & 15;                  // position for this A row
    const int m0   = blockIdx.x * 16;
    const int n    = m0 / HW;
    const int rem  = m0 % HW;
    const int h    = rem / Ww;
    const int w0   = rem % Ww;

    v8f acc0 = {}; v8f acc1 = {};
    const float* xn = x + (size_t)n * Cc * HW;

    for (int kk = 0; kk < KK; ++kk) {
        // --- bilinear params for position p (registers, branch-free) ---
        const size_t pb = (size_t)h * Ww + (w0 + p);
        const float oy = offsBuf[((size_t)(n * 18 + 2 * kk)) * HW + pb];
        const float ox = offsBuf[((size_t)(n * 18 + 2 * kk + 1)) * HW + pb];
        const float mv = maskBuf[((size_t)(n * 9 + kk)) * HW + pb];
        const float sy = (float)h + (float)(kk / 3 - 1) + oy;
        const float sx = (float)(w0 + p) + (float)(kk % 3 - 1) + ox;
        const float y0f = floorf(sy), x0f = floorf(sx);
        const int iy0 = (int)y0f, ix0 = (int)x0f;
        const int iy1 = iy0 + 1,  ix1 = ix0 + 1;
        const float wy1 = sy - y0f, wy0 = 1.f - wy1;
        const float wx1 = sx - x0f, wx0 = 1.f - wx1;
        const int cy0 = min(max(iy0, 0), Hh - 1), cy1 = min(max(iy1, 0), Hh - 1);
        const int cx0 = min(max(ix0, 0), Ww - 1), cx1 = min(max(ix1, 0), Ww - 1);
        const float vy0 = (iy0 >= 0 && iy0 < Hh) ? 1.f : 0.f;
        const float vy1 = (iy1 >= 0 && iy1 < Hh) ? 1.f : 0.f;
        const float vx0 = (ix0 >= 0 && ix0 < Ww) ? 1.f : 0.f;
        const float vx1 = (ix1 >= 0 && ix1 < Ww) ? 1.f : 0.f;
        const float g0 = mv * wy0 * wx0 * vy0 * vx0;  const int i0 = cy0 * Ww + cx0;
        const float g1 = mv * wy0 * wx1 * vy0 * vx1;  const int i1 = cy0 * Ww + cx1;
        const float g2 = mv * wy1 * wx0 * vy1 * vx0;  const int i2 = cy1 * Ww + cx0;
        const float g3 = mv * wy1 * wx1 * vy1 * vx1;  const int i3 = cy1 * Ww + cx1;

        for (int chunk = 0; chunk < 4; ++chunk) {
            __syncthreads();                  // prior A reads complete
            union { __bf16 b[4]; unsigned long long u64; } pk;
            #pragma unroll
            for (int je = 0; je < 4; ++je) {
                const int e = 4 * q + je;
                const int c = chunk * 32 + frag_kl_of(g, e);
                const float* xb = xn + (size_t)c * HW;
                pk.b[je] = (__bf16)(g0 * xb[i0] + g1 * xb[i1] + g2 * xb[i2] + g3 * xb[i3]);
            }
            *reinterpret_cast<unsigned long long*>(&As[l][4 * q]) = pk.u64;
            __syncthreads();                  // A tile visible to all waves
            const v16bf a  = *reinterpret_cast<const v16bf*>(&As[lane][0]);
            const int ktile = kk * 4 + chunk;
            const v16bf b0 = *reinterpret_cast<const v16bf*>(wB + ((size_t)(ktile * 8 + wav * 2 + 0) * 32 + lane) * 16);
            const v16bf b1 = *reinterpret_cast<const v16bf*>(wB + ((size_t)(ktile * 8 + wav * 2 + 1) * 32 + lane) * 16);
            acc0 = __builtin_amdgcn_wmma_f32_16x16x32_bf16(false, a, false, b0, (short)0, acc0, false, false);
            acc1 = __builtin_amdgcn_wmma_f32_16x16x32_bf16(false, a, false, b1, (short)0, acc1, false, false);
        }
    }

    // epilogue: bias + BN partial sums + stage to LDS for coalesced store
    const int rowHi = (lane >> 4) * 8;
    const int colLo = lane & 15;
    #pragma unroll
    for (int j = 0; j < 2; ++j) {
        const int o = wav * 32 + j * 16 + colLo;
        const float bo = bias[o];
        float s = 0.f, s2 = 0.f;
        const v8f acc = j ? acc1 : acc0;
        #pragma unroll
        for (int r = 0; r < 8; ++r) {
            const float v = acc[r] + bo;
            sOut[r + rowHi][o] = v;
            s += v; s2 += v * v;
        }
        atomicAdd(bnsum + o, s);
        atomicAdd(bnsumsq + o, s2);
    }
    __syncthreads();
    // coalesced write-out: 16 lanes cover 16 consecutive w for each o
    const int wI    = t & 15;
    const int oBase = t >> 4;                 // 0..7
    const size_t rowAddr = (size_t)h * Ww + w0 + wI;
    #pragma unroll
    for (int og = 0; og < 16; ++og) {
        const int o = og * 8 + oBase;
        out[((size_t)(n * CO + o)) * HW + rowAddr] = sOut[wI][o];
    }
}

// --------------------------------------------------- K5: BN finalize + ReLU
__global__ void k5_bn_relu(float* __restrict__ out,
                           const float* __restrict__ bnsum,
                           const float* __restrict__ bnsumsq,
                           const float* __restrict__ gamma,
                           const float* __restrict__ beta)
{
    const size_t idx = (size_t)blockIdx.x * blockDim.x + threadIdx.x;
    if (idx >= (size_t)Nb * CO * HW) return;
    const int o = (int)((idx / HW) % CO);
    const float inv = 1.f / (float)Mrows;
    const float mean = bnsum[o] * inv;
    const float var  = bnsumsq[o] * inv - mean * mean;
    const float sc   = rsqrtf(var + BN_EPS) * gamma[o];
    float v = (out[idx] - mean) * sc + beta[o];
    out[idx] = v > 0.f ? v : 0.f;
}

// ---------------------------------------------------------------------------
extern "C" void kernel_launch(void* const* d_in, const int* in_sizes, int n_in,
                              void* d_out, int out_size, void* d_ws, size_t ws_size,
                              hipStream_t stream) {
    const float* x     = (const float*)d_in[0];
    const float* offw  = (const float*)d_in[1];
    const float* offb  = (const float*)d_in[2];
    const float* modw  = (const float*)d_in[3];
    const float* modb  = (const float*)d_in[4];
    const float* wgt   = (const float*)d_in[5];
    const float* bias  = (const float*)d_in[6];
    const float* gamma = (const float*)d_in[7];
    const float* beta  = (const float*)d_in[8];
    float* out = (float*)d_out;

    // workspace carve-up (~4.35 MB total)
    char* ws = (char*)d_ws;
    float*  offsBuf = (float*)(ws);                                   // N*18*HW f32
    float*  maskBuf = (float*)(ws + (size_t)Nb * 18 * HW * 4);        // N*9*HW  f32
    float*  bnsum   = (float*)(ws + (size_t)Nb * 27 * HW * 4);        // 128 f32
    float*  bnsumsq = bnsum + 128;
    __bf16* wOM     = (__bf16*)((char*)(bnsumsq + 128));              // 36*2*32*16 bf16
    __bf16* wB      = wOM + (size_t)NKT * 2 * 32 * 16;                // 36*8*32*16 bf16

    k0_init_bn<<<1, 256, 0, stream>>>(bnsum, bnsumsq);
    k1_wcvt_main<<<(NKT * 8 * 32 + 255) / 256, 256, 0, stream>>>(wgt, wB);
    k2_wcvt_offmask<<<(NKT * 2 * 32 + 255) / 256, 256, 0, stream>>>(offw, modw, wOM);
    k3_offmask_gemm<<<Mrows / 16, 32, 0, stream>>>(x, wOM, offb, modb, offsBuf, maskBuf);
    k4_dcn_gemm<<<Mrows / 16, 128, 0, stream>>>(x, offsBuf, maskBuf, wB, bias, out, bnsum, bnsumsq);
    k5_bn_relu<<<(int)(((size_t)Nb * CO * HW + 255) / 256), 256, 0, stream>>>(out, bnsum, bnsumsq, gamma, beta);
}
